// AttnBlockpp_23845658427591
// MI455X (gfx1250) — matrix operand order repaired and checked by reading, not compiled, not run on hardware
//
#include <hip/hip_runtime.h>

typedef __bf16 bf16_t;
typedef __attribute__((ext_vector_type(16))) __bf16 v16bf;
typedef __attribute__((ext_vector_type(8)))  float  v8f;

union FragAB { uint4 q[2]; v16bf v; };

#define CCH 256      // channels
#define NSP 4096     // spatial positions (64*64)
#define NGRP 32      // groupnorm groups
#define NB 4         // batch

static __device__ __forceinline__ v8f wmma_bf16(const FragAB& a, const FragAB& b, v8f c) {
  // D = A(16x32 bf16) * B(32x16 bf16) + C(16x16 f32)
  return __builtin_amdgcn_wmma_f32_16x16x32_bf16(false, a.v, false, b.v, (short)0, c,
                                                 false, false);
}

// Async global->LDS 16-byte copy (ASYNCcnt path, ISA 08_async_tensor §4).
// LDS address = low 32 bits of the generic pointer (ISA §10.2 aperture rule).
static __device__ __forceinline__ void async_copy16(void* lds_dst, const void* gsrc) {
  unsigned l = (unsigned)(uintptr_t)lds_dst;
  unsigned long long g = (unsigned long long)(uintptr_t)gsrc;
  asm volatile("global_load_async_to_lds_b128 %0, %1, off" ::"v"(l), "v"(g) : "memory");
}
static __device__ __forceinline__ void async_wait0() {
  asm volatile("s_wait_asynccnt 0" ::: "memory");
}

// ---------------------------------------------------------------------------
// GroupNorm statistics: one block per (batch, group); group = 8 contiguous
// channels * 4096 spatial = 32768 contiguous floats.
// ---------------------------------------------------------------------------
__global__ __launch_bounds__(256) void gn_stats_kernel(const float* __restrict__ x,
                                                       float* __restrict__ mean,
                                                       float* __restrict__ rstd) {
  int bg = blockIdx.x;  // b*32 + g
  const float* xp = x + (size_t)bg * 8 * NSP;
  float s = 0.f, s2 = 0.f;
  for (int i = threadIdx.x; i < 8 * NSP; i += 256) {
    float v = xp[i];
    s += v;
    s2 += v * v;
  }
  __shared__ float rs[256], rq[256];
  rs[threadIdx.x] = s;
  rq[threadIdx.x] = s2;
  __syncthreads();
  for (int off = 128; off > 0; off >>= 1) {
    if (threadIdx.x < off) {
      rs[threadIdx.x] += rs[threadIdx.x + off];
      rq[threadIdx.x] += rq[threadIdx.x + off];
    }
    __syncthreads();
  }
  if (threadIdx.x == 0) {
    float m = rs[0] * (1.f / 32768.f);
    float v = rq[0] * (1.f / 32768.f) - m * m;
    mean[bg] = m;
    rstd[bg] = rsqrtf(v + 1e-6f);
  }
}

// ---------------------------------------------------------------------------
// GroupNorm apply + layout transpose [B,C,N] -> [B,N,C] (bf16) via LDS tile.
// ---------------------------------------------------------------------------
__global__ __launch_bounds__(256) void gn_apply_kernel(
    const float* __restrict__ x, const float* __restrict__ gamma,
    const float* __restrict__ beta, const float* __restrict__ mean,
    const float* __restrict__ rstd, bf16_t* __restrict__ h) {
  int b = blockIdx.y;
  int n0 = blockIdx.x * 64;
  __shared__ bf16_t tile[64][CCH + 10];
  int tid = threadIdx.x;
  for (int p = 0; p < 64; ++p) {
    int idx = p * 256 + tid;
    int c = idx >> 6, nn = idx & 63;  // coalesced read along n
    float v = x[((size_t)b * CCH + c) * NSP + n0 + nn];
    int bg = b * NGRP + (c >> 3);
    float xn = (v - mean[bg]) * rstd[bg] * gamma[c] + beta[c];
    tile[nn][c] = (bf16_t)xn;
  }
  __syncthreads();
  for (int nn = 0; nn < 64; ++nn)  // coalesced write along c
    h[((size_t)b * NSP + n0 + nn) * CCH + tid] = tile[nn][tid];
}

// ---------------------------------------------------------------------------
// Convert weights to bf16, stored TRANSPOSED: wbfT[which][d][c] = W[c][d].
// One-time scatter (0.5 MB), makes every GEMM B-tile load a b128 copy.
// ---------------------------------------------------------------------------
__global__ __launch_bounds__(256) void wconv_kernel(
    const float* __restrict__ w0, const float* __restrict__ w1,
    const float* __restrict__ w2, const float* __restrict__ w3,
    bf16_t* __restrict__ wbfT) {
  int i = blockIdx.x * 256 + threadIdx.x;  // grid covers 65536 exactly
  int c = i >> 8, d = i & 255;             // coalesced read along d
  int j = d * CCH + c;                     // transposed destination
  wbfT[j] = (bf16_t)w0[i];
  wbfT[65536 + j] = (bf16_t)w1[i];
  wbfT[2 * 65536 + j] = (bf16_t)w2[i];
  wbfT[3 * 65536 + j] = (bf16_t)w3[i];
}

// ---------------------------------------------------------------------------
// QKV GEMM: Out[b,n,d] = sum_c h[b,n,c]*W[c,d] + bias[d]; Q gets *1/sqrt(C).
// 128x128 block tile, 8 waves of 32x64, K-steps of 32, bf16 WMMA.
// V is written TRANSPOSED to [B,C,N] (via LDS stage) so flash attention can
// load its B-fragments with b128 copies instead of scalar transposes.
// ---------------------------------------------------------------------------
__global__ __launch_bounds__(256) void qkv_gemm_kernel(
    const bf16_t* __restrict__ h, const bf16_t* __restrict__ wbfT,
    const float* __restrict__ bq, const float* __restrict__ bk,
    const float* __restrict__ bv, bf16_t* __restrict__ Qo,
    bf16_t* __restrict__ Ko, bf16_t* __restrict__ Vt) {
  int zb = blockIdx.z;
  int b = zb / 3, which = zb % 3;
  const bf16_t* Wt = wbfT + (size_t)which * 65536;  // [d][c]
  const float* bias = (which == 0) ? bq : (which == 1) ? bk : bv;
  float oscale = (which == 0) ? 0.0625f : 1.0f;  // 256^-0.5 folded into Q
  int m0 = blockIdx.x * 128, n0 = blockIdx.y * 128;

  __shared__ __align__(16) bf16_t As[128][40];    // [m][k]
  __shared__ __align__(16) bf16_t Bs[128][40];    // [d][k] (W already transposed)
  __shared__ __align__(16) bf16_t Vst[128][136];  // V epilogue transpose stage

  int tid = threadIdx.x, w = tid >> 5, lane = tid & 31;
  int lh = lane >> 4, l16 = lane & 15;
  int wm = w >> 1, wn = w & 1;

  const v8f vz = {0.f, 0.f, 0.f, 0.f, 0.f, 0.f, 0.f, 0.f};
  v8f acc[2][4];
#pragma unroll
  for (int s = 0; s < 2; ++s)
#pragma unroll
    for (int t = 0; t < 4; ++t) acc[s][t] = vz;

  for (int k0 = 0; k0 < CCH; k0 += 32) {
    __syncthreads();
#pragma unroll
    for (int p = 0; p < 2; ++p) {  // A: 128x32 bf16, async b128
      int idx = p * 256 + tid;
      int row = idx >> 2, c4 = idx & 3;
      async_copy16(&As[row][c4 * 8],
                   &h[((size_t)b * NSP + m0 + row) * CCH + k0 + c4 * 8]);
    }
#pragma unroll
    for (int p = 0; p < 2; ++p) {  // B: 128x32 bf16 from W^T, async b128
      int idx = p * 256 + tid;
      int row = idx >> 2, c4 = idx & 3;
      async_copy16(&Bs[row][c4 * 8],
                   &Wt[(size_t)(n0 + row) * CCH + k0 + c4 * 8]);
    }
    async_wait0();
    __syncthreads();

    FragAB a[2];
#pragma unroll
    for (int s = 0; s < 2; ++s) {
      // A frag: lane row = l16; k = 16*lh + j
      const bf16_t* ap = &As[wm * 32 + s * 16 + l16][16 * lh];
      a[s].q[0] = *(const uint4*)ap;
      a[s].q[1] = *(const uint4*)(ap + 8);
    }
#pragma unroll
    for (int t = 0; t < 4; ++t) {
      // B frag: lane col = l16; k = 16*lh + j (contiguous in [n][k] layout)
      FragAB bb;
      const bf16_t* bp = &Bs[wn * 64 + t * 16 + l16][16 * lh];
      bb.q[0] = *(const uint4*)bp;
      bb.q[1] = *(const uint4*)(bp + 8);
#pragma unroll
      for (int s = 0; s < 2; ++s) acc[s][t] = wmma_bf16(a[s], bb, acc[s][t]);
    }
  }

  if (which < 2) {  // Q / K: direct [B,N,C] store
    bf16_t* Out = (which == 0) ? Qo : Ko;
#pragma unroll
    for (int s = 0; s < 2; ++s)
#pragma unroll
      for (int t = 0; t < 4; ++t) {
        int n = n0 + wn * 64 + t * 16 + l16;
        float bvv = bias[n];
#pragma unroll
        for (int r = 0; r < 8; ++r) {  // C/D frag: row = r + 8*lh, col = l16
          int m = m0 + wm * 32 + s * 16 + r + 8 * lh;
          Out[((size_t)b * NSP + m) * CCH + n] = (bf16_t)((acc[s][t][r] + bvv) * oscale);
        }
      }
  } else {  // V: transpose through LDS -> [B,C,N] coalesced store
#pragma unroll
    for (int s = 0; s < 2; ++s)
#pragma unroll
      for (int t = 0; t < 4; ++t) {
        int nl = wn * 64 + t * 16 + l16;
        float bvv = bias[n0 + nl];
#pragma unroll
        for (int r = 0; r < 8; ++r)
          Vst[wm * 32 + s * 16 + 8 * lh + r][nl] = (bf16_t)(acc[s][t][r] + bvv);
      }
    __syncthreads();
    for (int p = 0; p < 64; ++p) {
      int idx = p * 256 + tid;
      int d = idx >> 7, mm = idx & 127;
      Vt[((size_t)b * CCH + n0 + d) * NSP + m0 + mm] = Vst[mm][d];
    }
  }
}

// ---------------------------------------------------------------------------
// Flash attention: block = 128 queries (8 waves x 16 rows), key blocks of 64.
// S = Q*K^T (scale pre-folded), online softmax, O += P*V. N x N never stored.
// V arrives pre-transposed [B,C,N], so all tile loads are async b128 copies.
// ---------------------------------------------------------------------------
__global__ __launch_bounds__(256) void flash_attn_kernel(
    const bf16_t* __restrict__ Q, const bf16_t* __restrict__ K,
    const bf16_t* __restrict__ Vt, bf16_t* __restrict__ O) {
  int b = blockIdx.y;
  int q0 = blockIdx.x * 128;

  __shared__ __align__(16) bf16_t Qs[128][264];   // [query][c]
  __shared__ __align__(16) bf16_t Ks[64][264];    // [key][c] == B-layout for K^T
  __shared__ __align__(16) bf16_t Vs[256][72];    // [c][key] (from V^T, no transpose)
  __shared__ __align__(16) bf16_t Ps[8][16][72];  // per-wave P strip

  int tid = threadIdx.x, w = tid >> 5, lane = tid & 31;
  int lh = lane >> 4, l16 = lane & 15;

#pragma unroll 4
  for (int p = 0; p < 16; ++p) {  // Q tile once, async
    int idx = p * 256 + tid;
    int row = idx >> 5, c8 = idx & 31;
    async_copy16(&Qs[row][c8 * 8],
                 &Q[((size_t)b * NSP + q0 + row) * CCH + c8 * 8]);
  }

  const v8f vz = {0.f, 0.f, 0.f, 0.f, 0.f, 0.f, 0.f, 0.f};
  v8f Oacc[16];
#pragma unroll
  for (int nt = 0; nt < 16; ++nt) Oacc[nt] = vz;
  float mrow[8], lsum[8];
#pragma unroll
  for (int r = 0; r < 8; ++r) {
    mrow[r] = -3.0e38f;
    lsum[r] = 0.f;
  }

  for (int kb = 0; kb < NSP / 64; ++kb) {
    int k0 = kb * 64;
    __syncthreads();
#pragma unroll 2
    for (int p = 0; p < 8; ++p) {  // K block 64x256, async
      int idx = p * 256 + tid;
      int row = idx >> 5, c8 = idx & 31;
      async_copy16(&Ks[row][c8 * 8],
                   &K[((size_t)b * NSP + k0 + row) * CCH + c8 * 8]);
    }
#pragma unroll 2
    for (int p = 0; p < 8; ++p) {  // V block from V^T: rows are channels, async
      int idx = p * 256 + tid;
      int c = idx >> 3, k8 = idx & 7;
      async_copy16(&Vs[c][k8 * 8],
                   &Vt[((size_t)b * CCH + c) * NSP + k0 + k8 * 8]);
    }
    if (kb + 1 < NSP / 64)  // hint next K block toward GL2
      __builtin_prefetch(&K[((size_t)b * NSP + k0 + 64) * CCH] + (size_t)tid * 64, 0, 1);
    async_wait0();
    __syncthreads();

    // ---- S strip 16x64 = Q_strip * K_block^T ----
    v8f S[4];
#pragma unroll
    for (int t = 0; t < 4; ++t) S[t] = vz;
#pragma unroll
    for (int ks = 0; ks < 8; ++ks) {
      FragAB a;
      const bf16_t* ap = &Qs[w * 16 + l16][ks * 32 + 16 * lh];
      a.q[0] = *(const uint4*)ap;
      a.q[1] = *(const uint4*)(ap + 8);
#pragma unroll
      for (int t = 0; t < 4; ++t) {
        FragAB bb;
        const bf16_t* bp = &Ks[t * 16 + l16][ks * 32 + 16 * lh];
        bb.q[0] = *(const uint4*)bp;
        bb.q[1] = *(const uint4*)(bp + 8);
        S[t] = wmma_bf16(a, bb, S[t]);
      }
    }

    // ---- online softmax; row = r + 8*lh, reduce over 16 lanes of same half ----
#pragma unroll
    for (int r = 0; r < 8; ++r) {
      float mx = fmaxf(fmaxf(S[0][r], S[1][r]), fmaxf(S[2][r], S[3][r]));
#pragma unroll
      for (int d = 1; d < 16; d <<= 1) mx = fmaxf(mx, __shfl_xor(mx, d, 32));
      float mnew = fmaxf(mrow[r], mx);
      float alpha = __expf(mrow[r] - mnew);
      float ps = 0.f;
#pragma unroll
      for (int t = 0; t < 4; ++t) {
        float pv = __expf(S[t][r] - mnew);
        ps += pv;
        Ps[w][8 * lh + r][t * 16 + l16] = (bf16_t)pv;
      }
#pragma unroll
      for (int d = 1; d < 16; d <<= 1) ps += __shfl_xor(ps, d, 32);
      lsum[r] = lsum[r] * alpha + ps;
      mrow[r] = mnew;
#pragma unroll
      for (int nt = 0; nt < 16; ++nt) Oacc[nt][r] *= alpha;
    }

    // wave-local LDS RAW on Ps (cross-lane): drain DS counter before reload
    asm volatile("s_wait_dscnt 0" ::: "memory");

    // ---- O strip 16x256 += P(16x64) * V(64x256) ----
#pragma unroll
    for (int ks = 0; ks < 2; ++ks) {
      FragAB a;
      const bf16_t* ap = &Ps[w][l16][ks * 32 + 16 * lh];
      a.q[0] = *(const uint4*)ap;
      a.q[1] = *(const uint4*)(ap + 8);
#pragma unroll
      for (int nt = 0; nt < 16; ++nt) {
        FragAB bb;
        const bf16_t* vp = &Vs[nt * 16 + l16][ks * 32 + 16 * lh];
        bb.q[0] = *(const uint4*)vp;
        bb.q[1] = *(const uint4*)(vp + 8);
        Oacc[nt] = wmma_bf16(a, bb, Oacc[nt]);
      }
    }
  }

#pragma unroll
  for (int r = 0; r < 8; ++r) {
    float inv = 1.f / lsum[r];
    size_t rowoff = ((size_t)b * NSP + q0 + w * 16 + 8 * lh + r) * CCH;
#pragma unroll
    for (int nt = 0; nt < 16; ++nt)
      O[rowoff + nt * 16 + l16] = (bf16_t)(Oacc[nt][r] * inv);
  }
}

// ---------------------------------------------------------------------------
// Final nin + residual: out[b,d,n] = x[b,d,n] + b3[d] + sum_c O[b,n,c]*W3[c,d]
// GEMM like qkv; epilogue transposes through f32 LDS tile for coalesced store.
// ---------------------------------------------------------------------------
__global__ __launch_bounds__(256) void out_gemm_kernel(
    const bf16_t* __restrict__ Omat, const bf16_t* __restrict__ wbfT,
    const float* __restrict__ b3, const float* __restrict__ x,
    float* __restrict__ out) {
  int b = blockIdx.z;
  int m0 = blockIdx.x * 128, n0 = blockIdx.y * 128;
  const bf16_t* Wt = wbfT + (size_t)3 * 65536;  // [d][c]

  __shared__ __align__(16) bf16_t As[128][40];
  __shared__ __align__(16) bf16_t Bs[128][40];
  __shared__ float Cs[128][133];

  int tid = threadIdx.x, w = tid >> 5, lane = tid & 31;
  int lh = lane >> 4, l16 = lane & 15;
  int wm = w >> 1, wn = w & 1;

  const v8f vz = {0.f, 0.f, 0.f, 0.f, 0.f, 0.f, 0.f, 0.f};
  v8f acc[2][4];
#pragma unroll
  for (int s = 0; s < 2; ++s)
#pragma unroll
    for (int t = 0; t < 4; ++t) acc[s][t] = vz;

  for (int k0 = 0; k0 < CCH; k0 += 32) {
    __syncthreads();
#pragma unroll
    for (int p = 0; p < 2; ++p) {
      int idx = p * 256 + tid;
      int row = idx >> 2, c4 = idx & 3;
      async_copy16(&As[row][c4 * 8],
                   &Omat[((size_t)b * NSP + m0 + row) * CCH + k0 + c4 * 8]);
    }
#pragma unroll
    for (int p = 0; p < 2; ++p) {
      int idx = p * 256 + tid;
      int row = idx >> 2, c4 = idx & 3;
      async_copy16(&Bs[row][c4 * 8],
                   &Wt[(size_t)(n0 + row) * CCH + k0 + c4 * 8]);
    }
    async_wait0();
    __syncthreads();

    FragAB a[2];
#pragma unroll
    for (int s = 0; s < 2; ++s) {
      const bf16_t* ap = &As[wm * 32 + s * 16 + l16][16 * lh];
      a[s].q[0] = *(const uint4*)ap;
      a[s].q[1] = *(const uint4*)(ap + 8);
    }
#pragma unroll
    for (int t = 0; t < 4; ++t) {
      FragAB bb;
      const bf16_t* bp = &Bs[wn * 64 + t * 16 + l16][16 * lh];
      bb.q[0] = *(const uint4*)bp;
      bb.q[1] = *(const uint4*)(bp + 8);
#pragma unroll
      for (int s = 0; s < 2; ++s) acc[s][t] = wmma_bf16(a[s], bb, acc[s][t]);
    }
  }

#pragma unroll
  for (int s = 0; s < 2; ++s)
#pragma unroll
    for (int t = 0; t < 4; ++t) {
      int nl = wn * 64 + t * 16 + l16;
      float bvv = b3[n0 + nl];
#pragma unroll
      for (int r = 0; r < 8; ++r)
        Cs[wm * 32 + s * 16 + 8 * lh + r][nl] = acc[s][t][r] + bvv;
    }
  __syncthreads();
  for (int p = 0; p < 64; ++p) {  // transposed, coalesced f32 store + residual
    int idx = p * 256 + tid;
    int d = idx >> 7, mm = idx & 127;
    size_t o = ((size_t)b * CCH + n0 + d) * NSP + m0 + mm;
    out[o] = x[o] + Cs[mm][d];
  }
}

// ---------------------------------------------------------------------------
extern "C" void kernel_launch(void* const* d_in, const int* in_sizes, int n_in,
                              void* d_out, int out_size, void* d_ws, size_t ws_size,
                              hipStream_t stream) {
  const float* x = (const float*)d_in[0];
  const float* gamma = (const float*)d_in[1];
  const float* beta = (const float*)d_in[2];
  const float* W0 = (const float*)d_in[3];
  const float* b0 = (const float*)d_in[4];
  const float* W1 = (const float*)d_in[5];
  const float* b1 = (const float*)d_in[6];
  const float* W2 = (const float*)d_in[7];
  const float* b2 = (const float*)d_in[8];
  const float* W3 = (const float*)d_in[9];
  const float* b3 = (const float*)d_in[10];
  float* out = (float*)d_out;

  char* ws = (char*)d_ws;
  float* meanp = (float*)(ws);
  float* rstdp = (float*)(ws + 512);
  bf16_t* wbfT = (bf16_t*)(ws + 4096);
  size_t off = 4096 + (size_t)4 * 65536 * 2;
  const size_t tens = (size_t)NB * NSP * CCH * 2;  // 8 MB per bf16 tensor
  bf16_t* hbuf = (bf16_t*)(ws + off);
  off += tens;
  bf16_t* Qb = (bf16_t*)(ws + off);
  off += tens;
  bf16_t* Kb = (bf16_t*)(ws + off);
  off += tens;
  bf16_t* Vtb = (bf16_t*)(ws + off);
  bf16_t* Ob = hbuf;  // h is dead after the QKV GEMM; reuse for O

  wconv_kernel<<<256, 256, 0, stream>>>(W0, W1, W2, W3, wbfT);
  gn_stats_kernel<<<NB * NGRP, 256, 0, stream>>>(x, meanp, rstdp);
  gn_apply_kernel<<<dim3(NSP / 64, NB), 256, 0, stream>>>(x, gamma, beta, meanp, rstdp,
                                                          hbuf);
  qkv_gemm_kernel<<<dim3(NSP / 128, CCH / 128, NB * 3), 256, 0, stream>>>(
      hbuf, wbfT, b0, b1, b2, Qb, Kb, Vtb);
  flash_attn_kernel<<<dim3(NSP / 128, NB), 256, 0, stream>>>(Qb, Kb, Vtb, Ob);
  out_gemm_kernel<<<dim3(NSP / 128, CCH / 128, NB), 256, 0, stream>>>(Ob, wbfT, b3, x,
                                                                      out);
  (void)in_sizes;
  (void)n_in;
  (void)out_size;
  (void)ws_size;
}
